// Model_27857157882550
// MI455X (gfx1250) — matrix operand order repaired and checked by reading, not compiled, not run on hardware
//
#include <hip/hip_runtime.h>

// ---------------------------------------------------------------------------
// MI455X (gfx1250, wave32) implementation of the hetero-GNN forward pass.
// All GEMMs: v_wmma_f32_16x16x32_bf16 (bf16 in, f32 accumulate), with the
// B operand pre-packed as [N,K] so every fragment load is a contiguous
// 32-byte read.  Edge scatters: one wave32 per edge, coalesced f32 atomics.
// ---------------------------------------------------------------------------

typedef __bf16 bf16_t;
typedef __attribute__((ext_vector_type(16))) __bf16 v16bf;
typedef __attribute__((ext_vector_type(8)))  float  v8f;

struct Bf16x16 { uint4 a, b; };          // 32 bytes == one v16bf fragment

#define NDIS   8192
#define NDRUG  8192
#define NTOT   16384
#define IN_F   512
#define HID    128
#define HEADS  4
#define E_DD   250000
#define E_DR   500000
#define E_RD   500000
#define E_RR   250000
#define E_TOT  1500000

// ---------------------------------------------------------------------------
// Elementwise / conversion kernels
// ---------------------------------------------------------------------------
__global__ void k_f32_to_bf16(const float* __restrict__ s, bf16_t* __restrict__ d, long n) {
  long i = (long)blockIdx.x * blockDim.x + threadIdx.x;
  if (i < n) d[i] = (bf16_t)s[i];
}

// transpose-convert: s is [K,N] f32 row-major, d is [N,K] bf16 row-major
__global__ void k_f32_to_bf16_t(const float* __restrict__ s, bf16_t* __restrict__ d,
                                int K, int N) {
  long i = (long)blockIdx.x * blockDim.x + threadIdx.x;
  if (i >= (long)K * N) return;
  int nn = (int)(i / K);
  int kk = (int)(i % K);
  d[i] = (bf16_t)s[(long)kk * N + nn];
}

// dst[i] = bf16(src[i] * rowscale[i/128]); rowscale nullable
__global__ void k_scale_to_bf16(const float* __restrict__ s, const float* __restrict__ rs,
                                bf16_t* __restrict__ d, long n) {
  long i = (long)blockIdx.x * blockDim.x + threadIdx.x;
  if (i >= n) return;
  float v = s[i];
  if (rs) v *= rs[i >> 7];
  d[i] = (bf16_t)v;
}

__global__ void k_offset_copy(const int* __restrict__ s, int* __restrict__ d, int n, int off) {
  int i = blockIdx.x * blockDim.x + threadIdx.x;
  if (i < n) d[i] = s[i] + off;
}

__global__ void k_deg(const int* __restrict__ idx, float* __restrict__ deg, int E) {
  int i = blockIdx.x * blockDim.x + threadIdx.x;
  if (i < E) atomicAdd(deg + idx[i], 1.0f);
}

__global__ void k_rs_finalize(float* __restrict__ deg, long n) {
  long i = (long)blockIdx.x * blockDim.x + threadIdx.x;
  if (i < n) deg[i] = rsqrtf(fmaxf(deg[i], 1.0f));
}

__global__ void k_softmax4(const float* __restrict__ c, float* __restrict__ filt) {
  if (threadIdx.x == 0 && blockIdx.x == 0) {
    float m = fmaxf(fmaxf(c[0], c[1]), fmaxf(c[2], c[3]));
    float e0 = expf(c[0] - m), e1 = expf(c[1] - m), e2 = expf(c[2] - m), e3 = expf(c[3] - m);
    float s = e0 + e1 + e2 + e3;
    filt[0] = e0 / s; filt[1] = e1 / s; filt[2] = e2 / s; filt[3] = e3 / s;
  }
}

// H = relu(0.5*X + 0.5*A)   (FastGTN beta residual, BETA = 0.5)
__global__ void k_mix_relu(const float* __restrict__ x, const float* __restrict__ a,
                           float* __restrict__ h, long n) {
  long i = (long)blockIdx.x * blockDim.x + threadIdx.x;
  if (i < n) h[i] = fmaxf(0.5f * x[i] + 0.5f * a[i], 0.0f);
}

// ---------------------------------------------------------------------------
// Edge scatter: out[didx[e]][:] += coef * sscale[sidx[e]] * feat[sidx[e]][:]
// D = 128 floats per node; one wave32 per edge, lane covers 4 contiguous f32.
// ---------------------------------------------------------------------------
__global__ void k_scatter128(const float* __restrict__ feat, const int* __restrict__ sidx,
                             const int* __restrict__ didx, const float* __restrict__ sscale,
                             const float* __restrict__ coefp, float* __restrict__ outp, int E) {
  long t = (long)blockIdx.x * blockDim.x + threadIdx.x;
  int e = (int)(t >> 5);
  if (e >= E) return;
  int lane = (int)(t & 31);
  int s = sidx[e], d = didx[e];
  float sc = coefp ? coefp[0] : 1.0f;
  if (sscale) sc *= sscale[s];
  const float4 f = *reinterpret_cast<const float4*>(feat + (long)s * 128 + lane * 4);
  float* dst = outp + (long)d * 128 + lane * 4;
  atomicAdd(dst + 0, sc * f.x);
  atomicAdd(dst + 1, sc * f.y);
  atomicAdd(dst + 2, sc * f.z);
  atomicAdd(dst + 3, sc * f.w);
}

// ---------------------------------------------------------------------------
// BN column stats + apply (+ PReLU). X is [M,128].
// ---------------------------------------------------------------------------
__global__ void k_colstats(const float* __restrict__ X, int M,
                           float* __restrict__ mean, float* __restrict__ var) {
  int c = blockIdx.x;                       // 0..127
  float s = 0.f, s2 = 0.f;
  for (int r = threadIdx.x; r < M; r += blockDim.x) {
    float v = X[(long)r * 128 + c];
    s += v; s2 += v * v;
  }
  __shared__ float sh[256], sh2[256];
  sh[threadIdx.x] = s; sh2[threadIdx.x] = s2;
  __syncthreads();
  for (int st = 128; st > 0; st >>= 1) {
    if ((int)threadIdx.x < st) { sh[threadIdx.x] += sh[threadIdx.x + st]; sh2[threadIdx.x] += sh2[threadIdx.x + st]; }
    __syncthreads();
  }
  if (threadIdx.x == 0) {
    float mu = sh[0] / (float)M;
    mean[c] = mu;
    var[c] = sh2[0] / (float)M - mu * mu;
  }
}

__global__ void k_bn_prelu(const float* __restrict__ x, float* __restrict__ o,
                           const float* __restrict__ mean, const float* __restrict__ var,
                           const float* __restrict__ gamma, const float* __restrict__ beta,
                           const float* __restrict__ a, long n) {
  long i = (long)blockIdx.x * blockDim.x + threadIdx.x;
  if (i >= n) return;
  int c = (int)(i & 127);
  float y = (x[i] - mean[c]) * rsqrtf(var[c] + 1e-5f) * gamma[c] + beta[c];
  o[i] = (y >= 0.f) ? y : a[0] * y;
}

// ---------------------------------------------------------------------------
// GAT kernels
// ---------------------------------------------------------------------------
__global__ void k_el_er(const float* __restrict__ feat, const float* __restrict__ al,
                        const float* __restrict__ ar, float* __restrict__ el,
                        float* __restrict__ er, int n) {   // n = NTOT*HEADS
  int i = blockIdx.x * blockDim.x + threadIdx.x;
  if (i >= n) return;
  int node = i >> 2, h = i & 3;
  const float* f = feat + (long)node * 512 + h * 128;
  const float* A = al + h * 128;
  const float* B = ar + h * 128;
  float sl = 0.f, sr = 0.f;
  for (int d = 0; d < 128; ++d) { float v = f[d]; sl += v * A[d]; sr += v * B[d]; }
  el[i] = sl; er[i] = sr;
}

__global__ void k_edge_e(const int* __restrict__ u, const int* __restrict__ v,
                         const float* __restrict__ el, const float* __restrict__ er,
                         float* __restrict__ e, int n4) {  // n4 = E*HEADS
  int i = blockIdx.x * blockDim.x + threadIdx.x;
  if (i >= n4) return;
  int ed = i >> 2, h = i & 3;
  float x = el[u[ed] * 4 + h] + er[v[ed] * 4 + h];
  e[i] = (x >= 0.f) ? x : 0.2f * x;          // leaky_relu 0.2
}

__device__ __forceinline__ unsigned fkey(float f) {
  unsigned u = __float_as_uint(f);
  return (u & 0x80000000u) ? ~u : (u | 0x80000000u);
}
__device__ __forceinline__ float funkey(unsigned k) {
  unsigned u = (k & 0x80000000u) ? (k & 0x7fffffffu) : ~k;
  return __uint_as_float(u);
}

__global__ void k_seg_max(const float* __restrict__ e, const int* __restrict__ v,
                          unsigned* __restrict__ m, int n4) {
  int i = blockIdx.x * blockDim.x + threadIdx.x;
  if (i >= n4) return;
  atomicMax(m + v[i >> 2] * 4 + (i & 3), fkey(e[i]));
}

__global__ void k_edge_exp(float* __restrict__ e, const int* __restrict__ v,
                           const unsigned* __restrict__ m, float* __restrict__ s, int n4) {
  int i = blockIdx.x * blockDim.x + threadIdx.x;
  if (i >= n4) return;
  int vv = v[i >> 2] * 4 + (i & 3);
  float ex = expf(e[i] - funkey(m[vv]));
  e[i] = ex;                                  // e buffer now holds exp values
  atomicAdd(s + vv, ex);
}

// rst[v][h][:] += alpha(e,h) * feat[u][h][:]  — one wave per edge.
__global__ void k_gat_agg(const float* __restrict__ feat, const int* __restrict__ u,
                          const int* __restrict__ v, const float* __restrict__ ex,
                          const float* __restrict__ s, float* __restrict__ rst, int E) {
  long t = (long)blockIdx.x * blockDim.x + threadIdx.x;
  int e = (int)(t >> 5);
  if (e >= E) return;
  int lane = (int)(t & 31);
  int uu = u[e], vv = v[e];
  #pragma unroll
  for (int h = 0; h < 4; ++h) {
    float alpha = ex[(long)e * 4 + h] / fmaxf(s[vv * 4 + h], 1e-9f);
    const float4 f = *reinterpret_cast<const float4*>(feat + (long)uu * 512 + h * 128 + lane * 4);
    float* dst = rst + (long)vv * 512 + h * 128 + lane * 4;
    atomicAdd(dst + 0, alpha * f.x);
    atomicAdd(dst + 1, alpha * f.y);
    atomicAdd(dst + 2, alpha * f.z);
    atomicAdd(dst + 3, alpha * f.w);
  }
}

__global__ void k_gat_finish(const float* __restrict__ rst, const float* __restrict__ b,
                             const float* __restrict__ ga, const float* __restrict__ gaa,
                             float* __restrict__ nh, long n) {   // n = NTOT*128
  long i = (long)blockIdx.x * blockDim.x + threadIdx.x;
  if (i >= n) return;
  long node = i >> 7;
  int d = (int)(i & 127);
  float acc = 0.f;
  #pragma unroll
  for (int h = 0; h < 4; ++h) {
    float r = rst[node * 512 + h * 128 + d] + b[h * 128 + d];
    r = (r >= 0.f) ? r : ga[0] * r;
    acc += r;
  }
  acc *= 0.25f;                               // mean over heads
  nh[i] = (acc >= 0.f) ? acc : gaa[0] * acc;
}

// ---------------------------------------------------------------------------
// Semantic attention kernels
// ---------------------------------------------------------------------------
__global__ void k_stack_bf16(const float* __restrict__ s0, const float* __restrict__ s1,
                             const float* __restrict__ s2, const float* __restrict__ s3,
                             bf16_t* __restrict__ dst, long n) {   // n = rows*4*128
  long i = (long)blockIdx.x * blockDim.x + threadIdx.x;
  if (i >= n) return;
  long node = i >> 9;                        // /512
  int k = (int)((i >> 7) & 3);
  int d = (int)(i & 127);
  const float* s = (k == 0) ? s0 : (k == 1) ? s1 : (k == 2) ? s2 : s3;
  dst[i] = (bf16_t)s[node * 128 + d];
}

__global__ void k_semdot(const float* __restrict__ t, const float* __restrict__ W2,
                         float* __restrict__ w4, int rows) {      // wave per row
  long g = (long)blockIdx.x * blockDim.x + threadIdx.x;
  int r = (int)(g >> 5);
  if (r >= rows) return;
  int lane = (int)(g & 31);
  const float* row = t + (long)r * 128;
  float acc = 0.f;
  for (int j = lane; j < 128; j += 32) acc += row[j] * W2[j];
  for (int m = 16; m; m >>= 1) acc += __shfl_xor(acc, m, 32);
  if (lane == 0) atomicAdd(w4 + (r & 3), acc);
}

__global__ void k_sem_softmax(const float* __restrict__ w4, float* __restrict__ beta4, float invN) {
  if (threadIdx.x == 0 && blockIdx.x == 0) {
    float a0 = w4[0] * invN, a1 = w4[1] * invN, a2 = w4[2] * invN, a3 = w4[3] * invN;
    float m = fmaxf(fmaxf(a0, a1), fmaxf(a2, a3));
    float e0 = expf(a0 - m), e1 = expf(a1 - m), e2 = expf(a2 - m), e3 = expf(a3 - m);
    float s = e0 + e1 + e2 + e3;
    beta4[0] = e0 / s; beta4[1] = e1 / s; beta4[2] = e2 / s; beta4[3] = e3 / s;
  }
}

__global__ void k_sem_combine(const float* __restrict__ s0, const float* __restrict__ s1,
                              const float* __restrict__ s2, const float* __restrict__ s3,
                              const float* __restrict__ beta4, float* __restrict__ z, long n) {
  long i = (long)blockIdx.x * blockDim.x + threadIdx.x;
  if (i >= n) return;
  z[i] = beta4[0] * s0[i] + beta4[1] * s1[i] + beta4[2] * s2[i] + beta4[3] * s3[i];
}

// ---------------------------------------------------------------------------
// Unified WMMA GEMM: C[M,N] = epilogue(A[M,K] @ Bt[N,K]^T + bias)
//   A  : bf16 [M,K] row-major (activations)
//   Bt : bf16 [N,K] row-major (weights pre-packed N-major, or z_dis rows)
// 256 threads = 8 waves; wave w owns the 16x16 tile at column 16*w of a
// 16x128 output strip.  The 16x32 A tile is staged in LDS with 16B copies
// and shared by all 8 waves.  B fragment = one contiguous 32B read.
// act: 0=none 1=relu 2=tanh 3=sigmoid.  acc!=0: C += result.
// ---------------------------------------------------------------------------
__global__ __launch_bounds__(256) void k_gemm(
    const bf16_t* __restrict__ A, const bf16_t* __restrict__ Bt,
    const float* __restrict__ bias, float* __restrict__ C,
    int M, int N, int K, int act, int acc) {
  __shared__ __align__(16) bf16_t At[16 * 32];
  const int m0   = blockIdx.x * 16;
  const int n0   = blockIdx.y * 128;
  const int wave = threadIdx.x >> 5;
  const int lane = threadIdx.x & 31;
  const int hf   = lane >> 4;                // K-half selector
  const int l16  = lane & 15;
  const int col  = n0 + wave * 16 + l16;     // output column == row of Bt
  v8f cacc = {0.f, 0.f, 0.f, 0.f, 0.f, 0.f, 0.f, 0.f};
  for (int kk = 0; kk < K; kk += 32) {
    __syncthreads();
    if (threadIdx.x < 64) {                  // 64 x 16B = 16x32 bf16 tile
      int r = threadIdx.x >> 2, c8 = (threadIdx.x & 3) * 8;
      *reinterpret_cast<uint4*>(&At[r * 32 + c8]) =
          *reinterpret_cast<const uint4*>(&A[(long)(m0 + r) * K + kk + c8]);
    }
    __syncthreads();
    // A frag: row l16; elems 0..15 -> k = 16*hf+e
    Bf16x16 ar;
    ar.a = *reinterpret_cast<const uint4*>(&At[l16 * 32 + 16 * hf]);
    ar.b = *reinterpret_cast<const uint4*>(&At[l16 * 32 + 16 * hf + 8]);
    v16bf af = __builtin_bit_cast(v16bf, ar);
    // B frag: row `col` of Bt; k = kk + 16*hf + e  (32 contiguous bytes)
    Bf16x16 br = *reinterpret_cast<const Bf16x16*>(&Bt[(long)col * K + kk + 16 * hf]);
    v16bf bfr = __builtin_bit_cast(v16bf, br);
    cacc = __builtin_amdgcn_wmma_f32_16x16x32_bf16(false, af, false, bfr,
                                                   (short)0, cacc, false, false);
  }
  #pragma unroll
  for (int r = 0; r < 8; ++r) {
    int row = m0 + r + 8 * hf;
    long idx = (long)row * N + col;
    float v = cacc[r];
    if (bias) v += bias[col];
    if (acc)  v += C[idx];
    if (act == 1)      v = fmaxf(v, 0.f);
    else if (act == 2) v = tanhf(v);
    else if (act == 3) v = 1.0f / (1.0f + expf(-v));
    C[idx] = v;
  }
}

// ---------------------------------------------------------------------------
// Host orchestration
// ---------------------------------------------------------------------------
extern "C" void kernel_launch(void* const* d_in, const int* in_sizes, int n_in,
                              void* d_out, int out_size, void* d_ws, size_t ws_size,
                              hipStream_t stream) {
  (void)in_sizes; (void)n_in; (void)out_size; (void)ws_size;

  const float* x_dis    = (const float*)d_in[0];
  const float* x_drug   = (const float*)d_in[1];
  const float* W_dis    = (const float*)d_in[2];
  const float* b_dis    = (const float*)d_in[3];
  const float* W_drug   = (const float*)d_in[4];
  const float* b_drug   = (const float*)d_in[5];
  const float* gtn_W1   = (const float*)d_in[6];
  const float* gtn_b1   = (const float*)d_in[7];
  const float* gtn_conv = (const float*)d_in[8];
  const float* gtn_W2   = (const float*)d_in[9];
  const float* gtn_b2   = (const float*)d_in[10];
  const float* gcn1_W   = (const float*)d_in[11];
  const float* gcn1_b   = (const float*)d_in[12];
  const float* gcn1_gm  = (const float*)d_in[13];
  const float* gcn1_bt  = (const float*)d_in[14];
  const float* gcn1_a   = (const float*)d_in[15];
  const float* gcn2_W   = (const float*)d_in[16];
  const float* gcn2_b   = (const float*)d_in[17];
  const float* gcn2_gm  = (const float*)d_in[18];
  const float* gcn2_bt  = (const float*)d_in[19];
  const float* gcn2_a   = (const float*)d_in[20];
  const float* gat_W    = (const float*)d_in[21];
  const float* gat_al   = (const float*)d_in[22];
  const float* gat_ar   = (const float*)d_in[23];
  const float* gat_b    = (const float*)d_in[24];
  const float* gat_a    = (const float*)d_in[25];
  const float* ga_a     = (const float*)d_in[26];
  const float* sd_W1    = (const float*)d_in[27];
  const float* sd_b1    = (const float*)d_in[28];
  const float* sd_W2    = (const float*)d_in[29];
  const float* ss_W1    = (const float*)d_in[30];
  const float* ss_b1    = (const float*)d_in[31];
  const float* ss_W2    = (const float*)d_in[32];
  const float* dec_W    = (const float*)d_in[33];
  const int*   edd      = (const int*)d_in[34];   // [2][E_DD]
  const int*   edr      = (const int*)d_in[35];
  const int*   erd      = (const int*)d_in[36];
  const int*   err      = (const int*)d_in[37];
  float* out = (float*)d_out;

  char* wsb = (char*)d_ws;
  size_t off = 0;
  auto alloc = [&](size_t bytes) -> void* {
    off = (off + 255) & ~(size_t)255;
    void* p = wsb + off;
    off += bytes;
    return p;
  };
  dim3 blk(256);
  auto g1 = [](long n) { return dim3((unsigned)((n + 255) / 256)); };

  // ---- permanent allocations (all B operands packed [N,K]) --------------
  bf16_t* Wdis_b  = (bf16_t*)alloc((size_t)IN_F * HID * 2);    // [128,512]
  bf16_t* Wdrug_b = (bf16_t*)alloc((size_t)IN_F * HID * 2);
  bf16_t* gtnW1_b = (bf16_t*)alloc((size_t)HID * HID * 2);
  bf16_t* gtnW2_b = (bf16_t*)alloc((size_t)HID * HID * 2);
  bf16_t* gcn1W_b = (bf16_t*)alloc((size_t)4 * HID * HID * 2);
  bf16_t* gcn2W_b = (bf16_t*)alloc((size_t)4 * HID * HID * 2);
  bf16_t* gatW_b  = (bf16_t*)alloc((size_t)HID * 4 * HID * 2); // [512,128]
  bf16_t* sdW1_b  = (bf16_t*)alloc((size_t)HID * HID * 2);
  bf16_t* ssW1_b  = (bf16_t*)alloc((size_t)HID * HID * 2);
  bf16_t* decW_b  = (bf16_t*)alloc((size_t)HID * HID * 2);

  float* h_dis0  = (float*)alloc((size_t)NDIS * HID * 4);
  float* h_drug0 = (float*)alloc((size_t)NDRUG * HID * 4);
  float* Hgtn    = (float*)alloc((size_t)NTOT * HID * 4);
  float* h_dis2  = (float*)alloc((size_t)NDIS * HID * 4);
  float* h_drug2 = (float*)alloc((size_t)NDRUG * HID * 4);
  float* nh      = (float*)alloc((size_t)NTOT * HID * 4);

  float* rsbase  = (float*)alloc((size_t)8 * NDIS * 4);
  float* rs_dd_o = rsbase + 0 * NDIS; float* rs_dd_i = rsbase + 1 * NDIS;
  float* rs_dr_o = rsbase + 2 * NDIS; float* rs_dr_i = rsbase + 3 * NDIS;
  float* rs_rd_o = rsbase + 4 * NDIS; float* rs_rd_i = rsbase + 5 * NDIS;
  float* rs_rr_o = rsbase + 6 * NDIS; float* rs_rr_i = rsbase + 7 * NDIS;

  int* u_h = (int*)alloc((size_t)E_TOT * 4);
  int* v_h = (int*)alloc((size_t)E_TOT * 4);

  float* smalls = (float*)alloc(64 * 4);
  float* filt  = smalls;       // 4
  float* w4    = smalls + 8;   // 4
  float* beta4 = smalls + 16;  // 4

  float*  z_dis  = (float*)alloc((size_t)NDIS * HID * 4);
  float*  z_drug = (float*)alloc((size_t)NDRUG * HID * 4);
  float*  zd     = (float*)alloc((size_t)NDRUG * HID * 4);
  bf16_t* zdr_b  = (bf16_t*)alloc((size_t)NDRUG * HID * 2);
  bf16_t* zd_b   = (bf16_t*)alloc((size_t)NDRUG * HID * 2);
  bf16_t* zdis_b = (bf16_t*)alloc((size_t)NDIS * HID * 2);

  const size_t mark = off;

  // ---- phase 0: weight packing, homo edges, degrees, filt ---------------
  k_f32_to_bf16_t<<<g1((long)IN_F * HID), blk, 0, stream>>>(W_dis, Wdis_b, IN_F, HID);
  k_f32_to_bf16_t<<<g1((long)IN_F * HID), blk, 0, stream>>>(W_drug, Wdrug_b, IN_F, HID);
  k_f32_to_bf16_t<<<g1((long)HID * HID), blk, 0, stream>>>(gtn_W1, gtnW1_b, HID, HID);
  k_f32_to_bf16_t<<<g1((long)HID * HID), blk, 0, stream>>>(gtn_W2, gtnW2_b, HID, HID);
  for (int t = 0; t < 4; ++t) {
    k_f32_to_bf16_t<<<g1((long)HID * HID), blk, 0, stream>>>(
        gcn1_W + (size_t)t * HID * HID, gcn1W_b + (size_t)t * HID * HID, HID, HID);
    k_f32_to_bf16_t<<<g1((long)HID * HID), blk, 0, stream>>>(
        gcn2_W + (size_t)t * HID * HID, gcn2W_b + (size_t)t * HID * HID, HID, HID);
  }
  k_f32_to_bf16_t<<<g1((long)HID * 4 * HID), blk, 0, stream>>>(gat_W, gatW_b, HID, 4 * HID);
  k_f32_to_bf16_t<<<g1((long)HID * HID), blk, 0, stream>>>(sd_W1, sdW1_b, HID, HID);
  k_f32_to_bf16_t<<<g1((long)HID * HID), blk, 0, stream>>>(ss_W1, ssW1_b, HID, HID);
  k_f32_to_bf16_t<<<g1((long)HID * HID), blk, 0, stream>>>(dec_W, decW_b, HID, HID);

  // homogeneous edge list: dd | dr | rd | rr  (drug ids offset by NDIS)
  k_offset_copy<<<g1(E_DD), blk, 0, stream>>>(edd, u_h, E_DD, 0);
  k_offset_copy<<<g1(E_DR), blk, 0, stream>>>(edr, u_h + E_DD, E_DR, 0);
  k_offset_copy<<<g1(E_RD), blk, 0, stream>>>(erd, u_h + E_DD + E_DR, E_RD, NDIS);
  k_offset_copy<<<g1(E_RR), blk, 0, stream>>>(err, u_h + E_DD + E_DR + E_RD, E_RR, NDIS);
  k_offset_copy<<<g1(E_DD), blk, 0, stream>>>(edd + E_DD, v_h, E_DD, 0);
  k_offset_copy<<<g1(E_DR), blk, 0, stream>>>(edr + E_DR, v_h + E_DD, E_DR, NDIS);
  k_offset_copy<<<g1(E_RD), blk, 0, stream>>>(erd + E_RD, v_h + E_DD + E_DR, E_RD, 0);
  k_offset_copy<<<g1(E_RR), blk, 0, stream>>>(err + E_RR, v_h + E_DD + E_DR + E_RD, E_RR, NDIS);

  hipMemsetAsync(rsbase, 0, (size_t)8 * NDIS * 4, stream);
  k_deg<<<g1(E_DD), blk, 0, stream>>>(edd,        rs_dd_o, E_DD);
  k_deg<<<g1(E_DD), blk, 0, stream>>>(edd + E_DD, rs_dd_i, E_DD);
  k_deg<<<g1(E_DR), blk, 0, stream>>>(edr,        rs_dr_o, E_DR);
  k_deg<<<g1(E_DR), blk, 0, stream>>>(edr + E_DR, rs_dr_i, E_DR);
  k_deg<<<g1(E_RD), blk, 0, stream>>>(erd,        rs_rd_o, E_RD);
  k_deg<<<g1(E_RD), blk, 0, stream>>>(erd + E_RD, rs_rd_i, E_RD);
  k_deg<<<g1(E_RR), blk, 0, stream>>>(err,        rs_rr_o, E_RR);
  k_deg<<<g1(E_RR), blk, 0, stream>>>(err + E_RR, rs_rr_i, E_RR);
  k_rs_finalize<<<g1((long)8 * NDIS), blk, 0, stream>>>(rsbase, (long)8 * NDIS);

  k_softmax4<<<1, 1, 0, stream>>>(gtn_conv, filt);

  // ---- phase 1: input projections ---------------------------------------
  {
    off = mark;
    bf16_t* xdis_b  = (bf16_t*)alloc((size_t)NDIS * IN_F * 2);
    bf16_t* xdrug_b = (bf16_t*)alloc((size_t)NDRUG * IN_F * 2);
    k_f32_to_bf16<<<g1((long)NDIS * IN_F), blk, 0, stream>>>(x_dis, xdis_b, (long)NDIS * IN_F);
    k_f32_to_bf16<<<g1((long)NDRUG * IN_F), blk, 0, stream>>>(x_drug, xdrug_b, (long)NDRUG * IN_F);
    k_gemm<<<dim3(NDIS / 16, 1), blk, 0, stream>>>(xdis_b, Wdis_b, b_dis, h_dis0,
                                                   NDIS, HID, IN_F, 0, 0);
    k_gemm<<<dim3(NDRUG / 16, 1), blk, 0, stream>>>(xdrug_b, Wdrug_b, b_drug, h_drug0,
                                                    NDRUG, HID, IN_F, 0, 0);
  }

  // ---- phase 2: FastGTN --------------------------------------------------
  {
    off = mark;
    bf16_t* Xb   = (bf16_t*)alloc((size_t)NTOT * HID * 2);
    float*  Xf   = (float*)alloc((size_t)NTOT * HID * 4);
    float*  aggF = (float*)alloc((size_t)NTOT * HID * 4);
    bf16_t* Hb   = (bf16_t*)alloc((size_t)NTOT * HID * 2);

    k_f32_to_bf16<<<g1((long)NDIS * HID), blk, 0, stream>>>(h_dis0, Xb, (long)NDIS * HID);
    k_f32_to_bf16<<<g1((long)NDRUG * HID), blk, 0, stream>>>(h_drug0, Xb + (size_t)NDIS * HID,
                                                             (long)NDRUG * HID);
    k_gemm<<<dim3(NTOT / 16, 1), blk, 0, stream>>>(Xb, gtnW1_b, gtn_b1, Xf,
                                                   NTOT, HID, HID, 1, 0);   // relu
    hipMemsetAsync(aggF, 0, (size_t)NTOT * HID * 4, stream);
    const long segO[4] = {0, E_DD, E_DD + E_DR, E_DD + E_DR + E_RD};
    const int  segE[4] = {E_DD, E_DR, E_RD, E_RR};
    for (int t = 0; t < 4; ++t) {
      // out[u] += filt[t] * X_[v]
      k_scatter128<<<g1((long)segE[t] * 32), blk, 0, stream>>>(
          Xf, v_h + segO[t], u_h + segO[t], nullptr, filt + t, aggF, segE[t]);
    }
    k_mix_relu<<<g1((long)NTOT * HID), blk, 0, stream>>>(Xf, aggF, Xf, (long)NTOT * HID);
    k_f32_to_bf16<<<g1((long)NTOT * HID), blk, 0, stream>>>(Xf, Hb, (long)NTOT * HID);
    k_gemm<<<dim3(NTOT / 16, 1), blk, 0, stream>>>(Hb, gtnW2_b, gtn_b2, Hgtn,
                                                   NTOT, HID, HID, 0, 0);
  }

  // ---- phase 3: two HeteroGraphConv layers -------------------------------
  {
    off = mark;
    float*  hd1    = (float*)alloc((size_t)NDIS * HID * 4);
    float*  hr1    = (float*)alloc((size_t)NDRUG * HID * 4);
    float*  aggT   = (float*)alloc((size_t)NDIS * HID * 4);
    bf16_t* stageb = (bf16_t*)alloc((size_t)NDIS * HID * 2);
    float*  meanb  = (float*)alloc(HID * 4);
    float*  varb   = (float*)alloc(HID * 4);

    auto graphconv = [&](const float* hsrc, const int* src, const int* dst, int E,
                         const float* rs_o, const float* rs_i,
                         const bf16_t* Wb, const float* bias, float* outp, int acc) {
      hipMemsetAsync(aggT, 0, (size_t)NDIS * HID * 4, stream);
      k_scatter128<<<g1((long)E * 32), blk, 0, stream>>>(hsrc, src, dst, rs_o, nullptr, aggT, E);
      k_scale_to_bf16<<<g1((long)NDIS * HID), blk, 0, stream>>>(aggT, rs_i, stageb,
                                                                (long)NDIS * HID);
      k_gemm<<<dim3(NDIS / 16, 1), blk, 0, stream>>>(stageb, Wb, bias, outp,
                                                     NDIS, HID, HID, 0, acc);
    };
    auto hetero = [&](const float* hd, const float* hr, const bf16_t* Wb4, const float* b4,
                      const float* gm, const float* bt, const float* aP,
                      float* odn, float* orn) {
      graphconv(hd, edd,        edd + E_DD, E_DD, rs_dd_o, rs_dd_i,
                Wb4 + 0 * HID * HID, b4 + 0 * HID, odn, 0);
      graphconv(hr, erd,        erd + E_RD, E_RD, rs_rd_o, rs_rd_i,
                Wb4 + 2 * HID * HID, b4 + 2 * HID, odn, 1);
      graphconv(hd, edr,        edr + E_DR, E_DR, rs_dr_o, rs_dr_i,
                Wb4 + 1 * HID * HID, b4 + 1 * HID, orn, 0);
      graphconv(hr, err,        err + E_RR, E_RR, rs_rr_o, rs_rr_i,
                Wb4 + 3 * HID * HID, b4 + 3 * HID, orn, 1);
      k_colstats<<<dim3(HID), blk, 0, stream>>>(odn, NDIS, meanb, varb);
      k_bn_prelu<<<g1((long)NDIS * HID), blk, 0, stream>>>(odn, odn, meanb, varb,
                                                           gm, bt, aP, (long)NDIS * HID);
      k_colstats<<<dim3(HID), blk, 0, stream>>>(orn, NDRUG, meanb, varb);
      k_bn_prelu<<<g1((long)NDRUG * HID), blk, 0, stream>>>(orn, orn, meanb, varb,
                                                            gm, bt, aP, (long)NDRUG * HID);
    };
    hetero(h_dis0, h_drug0, gcn1W_b, gcn1_b, gcn1_gm, gcn1_bt, gcn1_a, hd1, hr1);
    hetero(hd1, hr1, gcn2W_b, gcn2_b, gcn2_gm, gcn2_bt, gcn2_a, h_dis2, h_drug2);
  }

  // ---- phase 4: GAT on homogeneous graph ---------------------------------
  {
    off = mark;
    bf16_t*   Xh_b = (bf16_t*)alloc((size_t)NTOT * HID * 2);
    float*    feat = (float*)alloc((size_t)NTOT * 512 * 4);
    float*    el   = (float*)alloc((size_t)NTOT * 4 * 4);
    float*    er   = (float*)alloc((size_t)NTOT * 4 * 4);
    unsigned* mmax = (unsigned*)alloc((size_t)NTOT * 4 * 4);
    float*    ssum = (float*)alloc((size_t)NTOT * 4 * 4);
    float*    ebuf = (float*)alloc((size_t)E_TOT * 4 * 4);
    float*    rst  = (float*)alloc((size_t)NTOT * 512 * 4);

    k_f32_to_bf16<<<g1((long)NDIS * HID), blk, 0, stream>>>(h_dis2, Xh_b, (long)NDIS * HID);
    k_f32_to_bf16<<<g1((long)NDRUG * HID), blk, 0, stream>>>(h_drug2, Xh_b + (size_t)NDIS * HID,
                                                             (long)NDRUG * HID);
    k_gemm<<<dim3(NTOT / 16, 4), blk, 0, stream>>>(Xh_b, gatW_b, nullptr, feat,
                                                   NTOT, 512, HID, 0, 0);
    k_el_er<<<g1((long)NTOT * 4), blk, 0, stream>>>(feat, gat_al, gat_ar, el, er, NTOT * 4);
    k_edge_e<<<g1((long)E_TOT * 4), blk, 0, stream>>>(u_h, v_h, el, er, ebuf, E_TOT * 4);
    hipMemsetAsync(mmax, 0, (size_t)NTOT * 4 * 4, stream);   // key 0 == -inf
    k_seg_max<<<g1((long)E_TOT * 4), blk, 0, stream>>>(ebuf, v_h, mmax, E_TOT * 4);
    hipMemsetAsync(ssum, 0, (size_t)NTOT * 4 * 4, stream);
    k_edge_exp<<<g1((long)E_TOT * 4), blk, 0, stream>>>(ebuf, v_h, mmax, ssum, E_TOT * 4);
    hipMemsetAsync(rst, 0, (size_t)NTOT * 512 * 4, stream);
    k_gat_agg<<<g1((long)E_TOT * 32), blk, 0, stream>>>(feat, u_h, v_h, ebuf, ssum, rst, E_TOT);
    k_gat_finish<<<g1((long)NTOT * HID), blk, 0, stream>>>(rst, gat_b, gat_a, ga_a, nh,
                                                           (long)NTOT * HID);
  }

  // ---- phase 5: semantic attention + decoder -----------------------------
  {
    off = mark;
    bf16_t* zstb = (bf16_t*)alloc((size_t)NDIS * 4 * HID * 2);
    float*  tbuf = (float*)alloc((size_t)NDIS * 4 * HID * 4);

    auto sematt = [&](const float* s0, const float* s1, const float* s2, const float* s3,
                      const bf16_t* W1b, const float* b1, const float* W2, float* z) {
      const long rows4 = (long)NDIS * 4;
      k_stack_bf16<<<g1(rows4 * HID), blk, 0, stream>>>(s0, s1, s2, s3, zstb, rows4 * HID);
      k_gemm<<<dim3((int)(rows4 / 16), 1), blk, 0, stream>>>(zstb, W1b, b1, tbuf,
                                                             (int)rows4, HID, HID, 2, 0); // tanh
      hipMemsetAsync(w4, 0, 4 * 4, stream);
      k_semdot<<<g1(rows4 * 32), blk, 0, stream>>>(tbuf, W2, w4, (int)rows4);
      k_sem_softmax<<<1, 1, 0, stream>>>(w4, beta4, 1.0f / (float)NDIS);
      k_sem_combine<<<g1((long)NDIS * HID), blk, 0, stream>>>(s0, s1, s2, s3, beta4, z,
                                                              (long)NDIS * HID);
    };
    // dis layers: [H_gtn[:NDIS], h_dis_initial, h_dis_after_gcn2, nh[:NDIS]]
    sematt(Hgtn, h_dis0, h_dis2, nh, ssW1_b, ss_b1, ss_W2, z_dis);
    // drug layers
    sematt(Hgtn + (size_t)NDIS * HID, h_drug0, h_drug2, nh + (size_t)NDIS * HID,
           sdW1_b, sd_b1, sd_W2, z_drug);

    // decoder: out = sigmoid((z_drug @ dec_W) @ z_dis^T)
    k_f32_to_bf16<<<g1((long)NDRUG * HID), blk, 0, stream>>>(z_drug, zdr_b, (long)NDRUG * HID);
    k_gemm<<<dim3(NDRUG / 16, 1), blk, 0, stream>>>(zdr_b, decW_b, nullptr, zd,
                                                    NDRUG, HID, HID, 0, 0);
    k_f32_to_bf16<<<g1((long)NDRUG * HID), blk, 0, stream>>>(zd, zd_b, (long)NDRUG * HID);
    k_f32_to_bf16<<<g1((long)NDIS * HID), blk, 0, stream>>>(z_dis, zdis_b, (long)NDIS * HID);
    // z_dis is [NDIS, HID] row-major == packed [N,K] for the NT product
    k_gemm<<<dim3(NDRUG / 16, NDIS / 128), blk, 0, stream>>>(zd_b, zdis_b, nullptr, out,
                                                             NDRUG, NDIS, HID, 3, 0); // sigmoid
  }
}
